// MultiHeadAttention_83202106458553
// MI455X (gfx1250) — matrix operand order repaired and checked by reading, not compiled, not run on hardware
//
#include <hip/hip_runtime.h>
#include <hip/hip_bf16.h>
#include <stdint.h>

#define DMODEL 768
#define NHEADS 12
#define DK     64
#define SEQ    2048
#define BATCH  4
#define BS     (BATCH*SEQ)   // 8192
#define KSTEPS (DMODEL/64)   // 12

typedef __bf16 bf16;
typedef __attribute__((ext_vector_type(16))) __bf16 v16bf;
typedef __attribute__((ext_vector_type(8)))  __bf16 v8bf;
typedef __attribute__((ext_vector_type(8)))  float  v8f;

union Frag16 { v16bf v; v8bf h[2]; };

static __device__ __forceinline__ v8f wmma_bf16(v16bf a, v16bf b, v8f c) {
  // D = A(16x32 bf16) x B(32x16 bf16) + C(16x16 f32)
  return __builtin_amdgcn_wmma_f32_16x16x32_bf16(false, a, false, b, (short)0, c, false, false);
}

// Async DMA of 16 bytes: global -> LDS, bypassing VGPRs (tracked by ASYNCcnt).
// LDS byte offset = low 32 bits of the generic LDS pointer.
static __device__ __forceinline__ void async_ld_b128(void* lds, const void* g) {
  asm volatile("global_load_async_to_lds_b128 %0, %1, off"
               :: "v"((uint32_t)(uintptr_t)lds), "v"((uint64_t)(uintptr_t)g)
               : "memory");
}
static __device__ __forceinline__ void wait_async0() {
  asm volatile("s_wait_asynccnt 0x0" ::: "memory");
}

// A fragment: 16x32 tile, row-major [row][k], leading dim ld.
// lane L: row = L&15, k-chunks at 8*(L>>4) and 16+8*(L>>4).
static __device__ __forceinline__ v16bf frag_A(const bf16* p, int ld, int lane) {
  int l15 = lane & 15, kg = lane >> 4;
  const bf16* r = p + l15 * ld + 8 * kg;
  Frag16 f;
  f.h[0] = *(const v8bf*)(r);
  f.h[1] = *(const v8bf*)(r + 16);
  return f.v;
}

// B fragment: B is 32x16 (KxN); stored transposed [n][k] row-major, leading dim ld.
// lane L: n = L&15, k-chunks at 8*(L>>4) and 16+8*(L>>4).
static __device__ __forceinline__ v16bf frag_Bt(const bf16* p, int ld, int lane) {
  int l15 = lane & 15, kg = lane >> 4;
  const bf16* r = p + l15 * ld + 8 * kg;
  Frag16 f;
  f.h[0] = *(const v8bf*)(r);
  f.h[1] = *(const v8bf*)(r + 16);
  return f.v;
}

// ---------------- prep kernels ----------------

__global__ void convert_x_kernel(const float* __restrict__ X, bf16* __restrict__ Xb, int n) {
  int i = blockIdx.x * blockDim.x + threadIdx.x;
  int stride = gridDim.x * blockDim.x;
  for (; i < n; i += stride) Xb[i] = (bf16)X[i];
}

// Wt[(mat*768 + o)*768 + i] = W_mat[i*768 + o]  (bf16)
__global__ void transpose_w_kernel(const float* __restrict__ Wq, const float* __restrict__ Wk,
                                   const float* __restrict__ Wv, const float* __restrict__ Wo,
                                   bf16* __restrict__ Wt) {
  int idx = blockIdx.x * 256 + threadIdx.x;        // 0 .. 4*768*768-1
  int mat = idx / (DMODEL * DMODEL);
  int rem = idx % (DMODEL * DMODEL);
  int o = rem / DMODEL, i = rem % DMODEL;
  const float* W = (mat == 0) ? Wq : (mat == 1) ? Wk : (mat == 2) ? Wv : Wo;
  Wt[idx] = (bf16)W[i * DMODEL + o];
}

// ---------------- QKV fused GEMM ----------------
// Y[8192 x 2304] = Xb[8192 x 768] @ [Wq|Wk|Wv] + bias; scatter into Q,K [B,H,S,Dk], Vt [B,H,Dk,S]
// Block tile 128x64, K-step 64, double-buffered async LDS staging, 8 wmma/stage/wave.

__global__ __launch_bounds__(256) void qkv_gemm_kernel(
    const bf16* __restrict__ X, const bf16* __restrict__ Wt,
    const float* __restrict__ bq, const float* __restrict__ bk, const float* __restrict__ bv,
    bf16* __restrict__ Q, bf16* __restrict__ K, bf16* __restrict__ Vt)
{
  __shared__ __align__(16) bf16 As[2][128 * 64];   // 2 x 16 KB
  __shared__ __align__(16) bf16 Bs[2][64 * 64];    // 2 x  8 KB
  const int t = threadIdx.x;
  const int lane = t & 31, wave = t >> 5;
  const int l15 = lane & 15, kg = lane >> 4;
  const int rowBase = blockIdx.y * 128;
  const int colBase = blockIdx.x * 64;             // within 2304
  const int mat = colBase / DMODEL;
  const int colLoc = colBase % DMODEL;

  // hoisted per-thread staging addresses (advance 64 elems per stage)
  int ofsA[4]; const bf16* pA[4];
#pragma unroll
  for (int it = 0; it < 4; ++it) {
    int c = t + it * 256;                          // 1024 chunks of 16B
    int row = c >> 3, off = (c & 7) * 8;
    ofsA[it] = row * 64 + off;
    pA[it] = &X[(size_t)(rowBase + row) * DMODEL + off];
  }
  int ofsB[2]; const bf16* pB[2];
#pragma unroll
  for (int it = 0; it < 2; ++it) {
    int c = t + it * 256;                          // 512 chunks of 16B
    int row = c >> 3, off = (c & 7) * 8;
    ofsB[it] = row * 64 + off;
    pB[it] = &Wt[(size_t)(mat * DMODEL + colLoc + row) * DMODEL + off];
  }

  auto issue = [&](int s) {
    int buf = s & 1;
#pragma unroll
    for (int it = 0; it < 4; ++it)
      async_ld_b128(&As[buf][ofsA[it]], pA[it] + s * 64);
#pragma unroll
    for (int it = 0; it < 2; ++it)
      async_ld_b128(&Bs[buf][ofsB[it]], pB[it] + s * 64);
  };

  v8f acc[4] = {};
  issue(0);
  for (int s = 0; s < KSTEPS; ++s) {
    wait_async0();                                 // this wave's stage-s tiles landed
    __syncthreads();                               // everyone's landed; prev reads done
    if (s + 1 < KSTEPS) issue(s + 1);              // DMA next stage under the MMA work
    if (s + 2 < KSTEPS)
      __builtin_prefetch(pA[0] + (s + 2) * 64, 0, 1);

    const bf16* Ab = As[s & 1];
    const bf16* Bb = Bs[s & 1];
    // load ALL fragments first (distinct registers) so ds waits don't serialize wmma
    v16bf a0 = frag_A(&Ab[(wave * 16) * 64 + 0], 64, lane);
    v16bf a1 = frag_A(&Ab[(wave * 16) * 64 + 32], 64, lane);
    v16bf bfr[2][4];
#pragma unroll
    for (int ks = 0; ks < 2; ++ks)
#pragma unroll
      for (int j = 0; j < 4; ++j)
        bfr[ks][j] = frag_Bt(&Bb[(j * 16) * 64 + ks * 32], 64, lane);
#pragma unroll
    for (int j = 0; j < 4; ++j) {
      acc[j] = wmma_bf16(a0, bfr[0][j], acc[j]);
      acc[j] = wmma_bf16(a1, bfr[1][j], acc[j]);
    }
  }

  const float* bias = (mat == 0) ? bq : (mat == 1) ? bk : bv;
#pragma unroll
  for (int j = 0; j < 4; ++j) {
    int col = colLoc + j * 16 + l15;               // 0..767
    float bb = bias[col];
    int h = col >> 6, d = col & 63;
#pragma unroll
    for (int r = 0; r < 8; ++r) {
      int row = rowBase + wave * 16 + r + 8 * kg;  // 0..8191
      int b = row >> 11, s = row & 2047;
      bf16 o = (bf16)(acc[j][r] + bb);
      if (mat == 2) {
        Vt[(((size_t)(b * NHEADS + h)) * DK + d) * SEQ + s] = o;
      } else {
        bf16* dst = (mat == 0) ? Q : K;
        dst[(((size_t)(b * NHEADS + h)) * SEQ + s) * DK + d] = o;
      }
    }
  }
}

// ---------------- flash attention ----------------
// grid.x = B*H, grid.y = S/128; 8 waves/block, each wave owns 16 query rows.
// Double-buffered async K/V^T staging; score matrix never touches HBM.

__global__ __launch_bounds__(256) void attn_kernel(
    const bf16* __restrict__ Q, const bf16* __restrict__ K,
    const bf16* __restrict__ Vt, bf16* __restrict__ Ctx)
{
  __shared__ __align__(16) bf16 Ks[2][32 * 64];    // [t][d]  (== B^T for Q@K^T)
  __shared__ __align__(16) bf16 Vs[2][64 * 32];    // [d][t]  (== B^T for P@V)
  __shared__ __align__(16) bf16 Ps[8][16 * 32];    // per-wave P tile

  const int t = threadIdx.x;
  const int lane = t & 31, wave = t >> 5;
  const int l15 = lane & 15, kg = lane >> 4;
  const int bh = blockIdx.x;
  const int b = bh / NHEADS, h = bh % NHEADS;
  const int qb = blockIdx.y * 128 + wave * 16;     // this wave's query-row base
  const size_t headSD = (size_t)bh * SEQ * DK;     // Q/K base
  const size_t headDS = (size_t)bh * DK * SEQ;     // Vt base

  // hoisted per-thread staging addresses
  const int rK = t >> 3, oK = (t & 7) * 8;         // K tile: 256 chunks
  const int rV = t >> 2, oV = (t & 3) * 8;         // V tile: 256 chunks
  const bf16* gK = &K[headSD + (size_t)rK * DK + oK];
  const bf16* gV = &Vt[headDS + (size_t)rV * SEQ + oV];

  auto issueKV = [&](int tt) {
    int buf = tt & 1;
    int t0 = tt * 32;
    async_ld_b128(&Ks[buf][rK * 64 + oK], gK + (size_t)t0 * DK);
    async_ld_b128(&Vs[buf][rV * 32 + oV], gV + t0);
  };

  // Q fragments (persist across the key loop)
  v16bf qf[2];
  {
    const bf16* qrow = Q + headSD + (size_t)(qb + l15) * DK;
#pragma unroll
    for (int ks = 0; ks < 2; ++ks) {
      Frag16 f;
      f.h[0] = *(const v8bf*)(qrow + ks * 32 + 8 * kg);
      f.h[1] = *(const v8bf*)(qrow + ks * 32 + 16 + 8 * kg);
      qf[ks] = f.v;
    }
  }

  v8f o0 = {}, o1 = {}, o2 = {}, o3 = {};
  float mrow[8], lrow[8];
#pragma unroll
  for (int r = 0; r < 8; ++r) { mrow[r] = -3.0e38f; lrow[r] = 0.0f; }

  const int ntBlock = (blockIdx.y * 128 + 128) / 32;   // key tiles the block needs
  issueKV(0);
  for (int tt = 0; tt < ntBlock; ++tt) {
    wait_async0();
    __syncthreads();
    if (tt + 1 < ntBlock) issueKV(tt + 1);         // DMA next tile under compute
    const int t0 = tt * 32;
    if (t0 > qb + 15) continue;                    // causal: nothing for this wave

    const bf16* Kb = Ks[tt & 1];
    const bf16* Vb = Vs[tt & 1];

    // S = Q @ K^T, two 16-col subtiles, K-dim 64; fragments loaded up-front
    v16bf k00 = frag_Bt(&Kb[0 * 64 + 0], 64, lane);
    v16bf k10 = frag_Bt(&Kb[16 * 64 + 0], 64, lane);
    v16bf k01 = frag_Bt(&Kb[0 * 64 + 32], 64, lane);
    v16bf k11 = frag_Bt(&Kb[16 * 64 + 32], 64, lane);
    v8f s0 = {}, s1 = {};
    s0 = wmma_bf16(qf[0], k00, s0);
    s1 = wmma_bf16(qf[0], k10, s1);
    s0 = wmma_bf16(qf[1], k01, s0);
    s1 = wmma_bf16(qf[1], k11, s1);

    // scale + causal mask + row max (16-lane half-wave reductions)
    const float scale = 0.125f;                    // 1/sqrt(64)
    float pmax[8];
#pragma unroll
    for (int r = 0; r < 8; ++r) {
      int sq = qb + r + 8 * kg;
      float v0 = s0[r] * scale, v1 = s1[r] * scale;
      if (t0 + l15 > sq)      v0 = -1e9f;
      if (t0 + 16 + l15 > sq) v1 = -1e9f;
      s0[r] = v0; s1[r] = v1;
      float mx = fmaxf(v0, v1);
      mx = fmaxf(mx, __shfl_xor(mx, 1));
      mx = fmaxf(mx, __shfl_xor(mx, 2));
      mx = fmaxf(mx, __shfl_xor(mx, 4));
      mx = fmaxf(mx, __shfl_xor(mx, 8));
      pmax[r] = mx;
    }

    // online softmax update; write P (bf16) into per-wave LDS for re-fragmenting
    float alpha[8];
#pragma unroll
    for (int r = 0; r < 8; ++r) {
      float mnew = fmaxf(mrow[r], pmax[r]);
      alpha[r] = __expf(mrow[r] - mnew);
      float p0 = __expf(s0[r] - mnew);
      float p1 = __expf(s1[r] - mnew);
      float psum = p0 + p1;
      psum += __shfl_xor(psum, 1);
      psum += __shfl_xor(psum, 2);
      psum += __shfl_xor(psum, 4);
      psum += __shfl_xor(psum, 8);
      lrow[r] = lrow[r] * alpha[r] + psum;
      mrow[r] = mnew;
      bf16* pp = &Ps[wave][(r + 8 * kg) * 32];
      pp[l15]      = (bf16)p0;
      pp[16 + l15] = (bf16)p1;
    }
    asm volatile("s_wait_dscnt 0x0" ::: "memory"); // wave-local LDS RAW

    v16bf ap = frag_A(&Ps[wave][0], 32, lane);
    v16bf vf0 = frag_Bt(&Vb[0 * 32], 32, lane);
    v16bf vf1 = frag_Bt(&Vb[16 * 32], 32, lane);
    v16bf vf2 = frag_Bt(&Vb[32 * 32], 32, lane);
    v16bf vf3 = frag_Bt(&Vb[48 * 32], 32, lane);
#pragma unroll
    for (int r = 0; r < 8; ++r) {
      o0[r] *= alpha[r]; o1[r] *= alpha[r]; o2[r] *= alpha[r]; o3[r] *= alpha[r];
    }
    o0 = wmma_bf16(ap, vf0, o0);
    o1 = wmma_bf16(ap, vf1, o1);
    o2 = wmma_bf16(ap, vf2, o2);
    o3 = wmma_bf16(ap, vf3, o3);
  }

  // normalize and store context (bf16) as [B*S, 768]
#pragma unroll
  for (int r = 0; r < 8; ++r) {
    float inv = 1.0f / lrow[r];
    int s = qb + r + 8 * kg;
    size_t rowOff = ((size_t)(b * SEQ + s)) * DMODEL + h * DK;
    Ctx[rowOff + 0 * 16 + l15] = (bf16)(o0[r] * inv);
    Ctx[rowOff + 1 * 16 + l15] = (bf16)(o1[r] * inv);
    Ctx[rowOff + 2 * 16 + l15] = (bf16)(o2[r] * inv);
    Ctx[rowOff + 3 * 16 + l15] = (bf16)(o3[r] * inv);
  }
}

// ---------------- output projection ----------------
// Out[8192 x 768] (f32) = Ctx[8192 x 768] @ Wo + bo. Same tiling as QKV GEMM.

__global__ __launch_bounds__(256) void oproj_gemm_kernel(
    const bf16* __restrict__ Ctx, const bf16* __restrict__ Wot,
    const float* __restrict__ bo, float* __restrict__ Out)
{
  __shared__ __align__(16) bf16 As[2][128 * 64];
  __shared__ __align__(16) bf16 Bs[2][64 * 64];
  const int t = threadIdx.x;
  const int lane = t & 31, wave = t >> 5;
  const int l15 = lane & 15, kg = lane >> 4;
  const int rowBase = blockIdx.y * 128;
  const int colBase = blockIdx.x * 64;

  int ofsA[4]; const bf16* pA[4];
#pragma unroll
  for (int it = 0; it < 4; ++it) {
    int c = t + it * 256;
    int row = c >> 3, off = (c & 7) * 8;
    ofsA[it] = row * 64 + off;
    pA[it] = &Ctx[(size_t)(rowBase + row) * DMODEL + off];
  }
  int ofsB[2]; const bf16* pB[2];
#pragma unroll
  for (int it = 0; it < 2; ++it) {
    int c = t + it * 256;
    int row = c >> 3, off = (c & 7) * 8;
    ofsB[it] = row * 64 + off;
    pB[it] = &Wot[(size_t)(colBase + row) * DMODEL + off];
  }

  auto issue = [&](int s) {
    int buf = s & 1;
#pragma unroll
    for (int it = 0; it < 4; ++it)
      async_ld_b128(&As[buf][ofsA[it]], pA[it] + s * 64);
#pragma unroll
    for (int it = 0; it < 2; ++it)
      async_ld_b128(&Bs[buf][ofsB[it]], pB[it] + s * 64);
  };

  v8f acc[4] = {};
  issue(0);
  for (int s = 0; s < KSTEPS; ++s) {
    wait_async0();
    __syncthreads();
    if (s + 1 < KSTEPS) issue(s + 1);
    if (s + 2 < KSTEPS)
      __builtin_prefetch(pA[0] + (s + 2) * 64, 0, 1);

    const bf16* Ab = As[s & 1];
    const bf16* Bb = Bs[s & 1];
    v16bf a0 = frag_A(&Ab[(wave * 16) * 64 + 0], 64, lane);
    v16bf a1 = frag_A(&Ab[(wave * 16) * 64 + 32], 64, lane);
    v16bf bfr[2][4];
#pragma unroll
    for (int ks = 0; ks < 2; ++ks)
#pragma unroll
      for (int j = 0; j < 4; ++j)
        bfr[ks][j] = frag_Bt(&Bb[(j * 16) * 64 + ks * 32], 64, lane);
#pragma unroll
    for (int j = 0; j < 4; ++j) {
      acc[j] = wmma_bf16(a0, bfr[0][j], acc[j]);
      acc[j] = wmma_bf16(a1, bfr[1][j], acc[j]);
    }
  }

#pragma unroll
  for (int j = 0; j < 4; ++j) {
    int col = colBase + j * 16 + l15;
    float bb = bo[col];
#pragma unroll
    for (int r = 0; r < 8; ++r) {
      int row = rowBase + wave * 16 + r + 8 * kg;
      Out[(size_t)row * DMODEL + col] = acc[j][r] + bb;
    }
  }
}

// ---------------- launch ----------------

extern "C" void kernel_launch(void* const* d_in, const int* in_sizes, int n_in,
                              void* d_out, int out_size, void* d_ws, size_t ws_size,
                              hipStream_t stream) {
  (void)in_sizes; (void)n_in; (void)out_size; (void)ws_size;
  const float* x  = (const float*)d_in[0];
  // d_in[1] = mask (causal tril) — computed analytically in-kernel
  const float* Wq = (const float*)d_in[2];
  const float* bq = (const float*)d_in[3];
  const float* Wk = (const float*)d_in[4];
  const float* bk = (const float*)d_in[5];
  const float* Wv = (const float*)d_in[6];
  const float* bv = (const float*)d_in[7];
  const float* Wo = (const float*)d_in[8];
  const float* bo = (const float*)d_in[9];
  float* out = (float*)d_out;

  const size_t N_X = (size_t)BS * DMODEL;          // 6,291,456 elems
  const size_t N_W = (size_t)4 * DMODEL * DMODEL;  // 2,359,296 elems
  char* ws = (char*)d_ws;
  bf16* Xb  = (bf16*)ws;
  bf16* Wt  = (bf16*)(ws + N_X * 2);
  bf16* Qb  = (bf16*)(ws + N_X * 2 + N_W * 2);
  bf16* Kb  = Qb + N_X;
  bf16* Vtb = Kb + N_X;
  bf16* Ctx = Xb;                                   // alias: x_bf dead after QKV GEMM

  convert_x_kernel<<<6144, 256, 0, stream>>>(x, Xb, (int)N_X);
  transpose_w_kernel<<<(int)(N_W / 256), 256, 0, stream>>>(Wq, Wk, Wv, Wo, Wt);
  qkv_gemm_kernel<<<dim3(36, 64), 256, 0, stream>>>(Xb, Wt, bq, bk, bv, Qb, Kb, Vtb);
  attn_kernel<<<dim3(BATCH * NHEADS, SEQ / 128), 256, 0, stream>>>(Qb, Kb, Vtb, Ctx);
  oproj_gemm_kernel<<<dim3(12, 64), 256, 0, stream>>>(Ctx, Wt + (size_t)3 * DMODEL * DMODEL, bo, out);
}
